// ModalityEncoder_17239998726184
// MI455X (gfx1250) — matrix operand order repaired and checked by reading, not compiled, not run on hardware
//
#include <hip/hip_runtime.h>
#include <hip/hip_bf16.h>

// ---------------------------------------------------------------------------
// Problem constants (match reference)
// ---------------------------------------------------------------------------
static constexpr int Bz   = 4;
static constexpr int LIN  = 2048;
static constexpr int DIN  = 512;
static constexpr int D    = 1024;
static constexpr int NQ   = 256;
static constexpr int H    = 16;
static constexpr int HD   = 64;         // D / H
static constexpr int NB   = 4;
static constexpr int DI   = 2048;
static constexpr int DS   = 16;
static constexpr int KCONV= 4;
static constexpr int DTR  = 64;
static constexpr int XPN  = DTR + 2*DS; // 96

// ---------------------------------------------------------------------------
// WMMA types (CDNA5 gfx1250, wave32)
// ---------------------------------------------------------------------------
typedef __bf16 v16bf __attribute__((ext_vector_type(16)));
typedef float  v8f   __attribute__((ext_vector_type(8)));

union ABFrag { v16bf v; uint4 q[2]; };

__device__ __forceinline__ unsigned short f2bf(float f) {
    unsigned int u = __float_as_uint(f);
    u += 0x7FFFu + ((u >> 16) & 1u);       // round-to-nearest-even
    return (unsigned short)(u >> 16);
}
__device__ __forceinline__ int imin(int a, int b) { return a < b ? a : b; }

// Issue one 16-byte async global->LDS transfer (ASYNCcnt-tracked).
__device__ __forceinline__ void async_b128(const unsigned short* g, unsigned short* l) {
    unsigned long long ga = (unsigned long long)g;
    unsigned int       la = (unsigned int)(unsigned long long)l;   // LDS address taken from the low 32 bits
    asm volatile("global_load_async_to_lds_b128 %0, %1, off" :: "v"(la), "v"(ga) : "memory");
}
__device__ __forceinline__ void wait_async0() {
    asm volatile("s_wait_asynccnt 0x0" ::: "memory");
}

// ---------------------------------------------------------------------------
// bf16 WMMA GEMM:  out(M,N) = A(M,K) @ W(N,K)^T [+ bias] [+ out_old]
// 8 waves/block arranged 2(M) x 4(N); wave tile 32x64 (2x4 fragments,
// 8 WMMAs per K-step); block tile 64(M) x 256(N). K-tiles of 32 are staged
// into LDS with global_load_async_to_lds_b128 (double buffered, ASYNCcnt).
// A/B fragments use contiguous k = 16*half + s (same order for both); C/D uses the ISA layout.
// ---------------------------------------------------------------------------
__global__ __launch_bounds__(256)
void k_gemm_bf16(const unsigned short* __restrict__ A,
                 const unsigned short* __restrict__ W,
                 const float* __restrict__ bias,
                 float* __restrict__ outF,
                 unsigned short* __restrict__ outB,
                 int M, int N, int K, int accF)
{
    __shared__ unsigned short tA[2][64 * 32];    //  8 KB
    __shared__ unsigned short tB[2][256 * 32];   // 32 KB

    const int tid  = threadIdx.x;
    const int wid  = tid >> 5;
    const int lane = tid & 31;
    const int half = lane >> 4;
    const int l15  = lane & 15;

    const int bm = blockIdx.y * 64;
    const int bn = blockIdx.x * 256;
    const int wm = (wid >> 2) * 32;              // wave tile origin inside block
    const int wn = (wid & 3) * 64;

    const int koffA = half ? 16 : 0;             // A frag: K 0..15 / 16..31 per half-wave
    const int koffB = half ? 16 : 0;             // B frag: K 0..15 / 16..31 per half-wave

    // cooperative stage mapping: A = 64 rows x 4 chunks (1/thread);
    //                            B = 256 rows x 4 chunks (4/thread)
    const int arow = tid >> 2, ach = (tid & 3) * 8;
    const int garow = imin(bm + arow, M - 1);

    const bool fullN = (bn + 256) <= N;          // block-uniform fast path
    bool bv[4];
    #pragma unroll
    for (int j = 0; j < 4; ++j) bv[j] = (bn + wn + j * 16 + l15) < N;
    const uint4 z4 = {0u, 0u, 0u, 0u};

    v8f c[2][4] = {};
    const int nk = K >> 5;

    // prologue: stage k-tile 0 into buffer 0
    {
        async_b128(A + (size_t)garow * K + ach, &tA[0][arow * 32 + ach]);
        #pragma unroll
        for (int s = 0; s < 4; ++s) {
            int idx = tid + s * 256;
            int br = idx >> 2, bc = (idx & 3) * 8;
            async_b128(W + (size_t)imin(bn + br, N - 1) * K + bc, &tB[0][br * 32 + bc]);
        }
    }

    for (int ki = 0; ki < nk; ++ki) {
        const int buf = ki & 1;
        wait_async0();                 // my stage-ki transfers have landed in LDS
        __syncthreads();               // everyone's have; prev buffer free for reuse

        if (ki + 1 < nk) {             // overlap next-tile fetch with this tile's WMMAs
            const int kk = (ki + 1) << 5;
            async_b128(A + (size_t)garow * K + kk + ach, &tA[buf ^ 1][arow * 32 + ach]);
            #pragma unroll
            for (int s = 0; s < 4; ++s) {
                int idx = tid + s * 256;
                int br = idx >> 2, bc = (idx & 3) * 8;
                async_b128(W + (size_t)imin(bn + br, N - 1) * K + kk + bc,
                           &tB[buf ^ 1][br * 32 + bc]);
            }
        }

        const unsigned short* sa = tA[buf];
        const unsigned short* sb = tB[buf];
        ABFrag a0, a1, b[4];
        const uint4* pa0 = (const uint4*)(sa + (wm + l15) * 32 + koffA);
        const uint4* pa1 = (const uint4*)(sa + (wm + 16 + l15) * 32 + koffA);
        a0.q[0] = pa0[0]; a0.q[1] = pa0[1];      // K: koffA..+15
        a1.q[0] = pa1[0]; a1.q[1] = pa1[1];
        if (fullN) {
            #pragma unroll
            for (int j = 0; j < 4; ++j) {
                const uint4* pb = (const uint4*)(sb + (wn + j * 16 + l15) * 32 + koffB);
                b[j].q[0] = pb[0]; b[j].q[1] = pb[1];
            }
        } else {
            #pragma unroll
            for (int j = 0; j < 4; ++j) {
                if (bv[j]) {
                    const uint4* pb = (const uint4*)(sb + (wn + j * 16 + l15) * 32 + koffB);
                    b[j].q[0] = pb[0]; b[j].q[1] = pb[1];
                } else { b[j].q[0] = z4; b[j].q[1] = z4; }
            }
        }

        #pragma unroll
        for (int j = 0; j < 4; ++j) {
            c[0][j] = __builtin_amdgcn_wmma_f32_16x16x32_bf16(false, a0.v, false, b[j].v, (short)0, c[0][j], false, false);
            c[1][j] = __builtin_amdgcn_wmma_f32_16x16x32_bf16(false, a1.v, false, b[j].v, (short)0, c[1][j], false, false);
        }
    }

    // C/D layout: VGPR r -> row r (lanes 0-15) or r+8 (lanes 16-31); col = l15.
    #pragma unroll
    for (int i = 0; i < 2; ++i)
        #pragma unroll
        for (int j = 0; j < 4; ++j)
            #pragma unroll
            for (int r = 0; r < 8; ++r) {
                int row = bm + wm + i * 16 + r + half * 8;
                int col = bn + wn + j * 16 + l15;
                if (row < M && col < N) {
                    size_t idx = (size_t)row * N + col;
                    float v = c[i][j][r];
                    if (bias) v += bias[col];
                    if (accF) v += outF[idx];
                    if (outF) outF[idx] = v;
                    if (outB) outB[idx] = f2bf(v);
                }
            }
}

// ---------------------------------------------------------------------------
// Flash attention over latents: grid = B*H blocks, 8 waves x 32 queries each.
// qh is batch-invariant (NQ,D). kh: (B*LIN, D). vT: (B,H,HD,LIN). Output bf16.
// ---------------------------------------------------------------------------
__global__ __launch_bounds__(256)
void k_flash_attn(const unsigned short* __restrict__ qh,
                  const unsigned short* __restrict__ kh,
                  const unsigned short* __restrict__ vT,
                  const unsigned char*  __restrict__ mask,
                  unsigned short* __restrict__ oat)
{
    __shared__ unsigned short ldsP[8 * 32 * 32];   // one 32x32 P tile per wave

    const int bh  = blockIdx.x;
    const int b   = bh >> 4;
    const int hh  = bh & 15;
    const int wid = threadIdx.x >> 5;
    const int lane = threadIdx.x & 31;
    const int half = lane >> 4;
    const int l15  = lane & 15;
    const int q0   = wid * 32;

    const int koffA = half ? 16 : 0;
    const int koffB = half ? 16 : 0;

    // Load Q fragments once (2 row-halves x 2 k-chunks of 32 over HD=64)
    ABFrag qf[2][2];
    #pragma unroll
    for (int i = 0; i < 2; ++i)
        #pragma unroll
        for (int kc = 0; kc < 2; ++kc) {
            const uint4* p = (const uint4*)(qh + (size_t)(q0 + i * 16 + l15) * D
                                               + hh * HD + kc * 32 + koffA);
            qf[i][kc].q[0] = p[0]; qf[i][kc].q[1] = p[1];
        }

    v8f o[2][4] = {};
    float mrow[2][8], lrow[2][8];
    #pragma unroll
    for (int i = 0; i < 2; ++i)
        #pragma unroll
        for (int r = 0; r < 8; ++r) { mrow[i][r] = -3.0e38f; lrow[i][r] = 0.f; }

    for (int l0 = 0; l0 < LIN; l0 += 32) {
        if (l0 + 32 < LIN)
            __builtin_prefetch(kh + (size_t)(b * LIN + l0 + 32 + l15) * D + hh * HD, 0, 3);
        v8f s[2][2] = {};
        #pragma unroll
        for (int kc = 0; kc < 2; ++kc) {
            ABFrag kf0, kf1;
            const uint4* p0 = (const uint4*)(kh + (size_t)(b * LIN + l0 + l15) * D
                                                + hh * HD + kc * 32 + koffB);
            const uint4* p1 = (const uint4*)(kh + (size_t)(b * LIN + l0 + 16 + l15) * D
                                                + hh * HD + kc * 32 + koffB);
            kf0.q[0] = p0[0]; kf0.q[1] = p0[1];
            kf1.q[0] = p1[0]; kf1.q[1] = p1[1];
            s[0][0] = __builtin_amdgcn_wmma_f32_16x16x32_bf16(false, qf[0][kc].v, false, kf0.v, (short)0, s[0][0], false, false);
            s[0][1] = __builtin_amdgcn_wmma_f32_16x16x32_bf16(false, qf[0][kc].v, false, kf1.v, (short)0, s[0][1], false, false);
            s[1][0] = __builtin_amdgcn_wmma_f32_16x16x32_bf16(false, qf[1][kc].v, false, kf0.v, (short)0, s[1][0], false, false);
            s[1][1] = __builtin_amdgcn_wmma_f32_16x16x32_bf16(false, qf[1][kc].v, false, kf1.v, (short)0, s[1][1], false, false);
        }

        // scale + key mask (mask depends only on column/key)
        unsigned char mk[2];
        mk[0] = mask[b * LIN + l0 + l15];
        mk[1] = mask[b * LIN + l0 + 16 + l15];
        #pragma unroll
        for (int i = 0; i < 2; ++i)
            #pragma unroll
            for (int j = 0; j < 2; ++j)
                #pragma unroll
                for (int r = 0; r < 8; ++r) {
                    float v = s[i][j][r] * 0.125f;      // 1/sqrt(64)
                    s[i][j][r] = mk[j] ? v : -1.0e9f;
                }

        // online softmax; rows this lane touches are consistent within half-wave
        #pragma unroll
        for (int i = 0; i < 2; ++i)
            #pragma unroll
            for (int r = 0; r < 8; ++r) {
                float rm = fmaxf(s[i][0][r], s[i][1][r]);
                #pragma unroll
                for (int off = 1; off < 16; off <<= 1)
                    rm = fmaxf(rm, __shfl_xor(rm, off, 32));
                float mnew = fmaxf(mrow[i][r], rm);
                float corr = __expf(mrow[i][r] - mnew);
                mrow[i][r] = mnew;
                float p0 = __expf(s[i][0][r] - mnew);
                float p1 = __expf(s[i][1][r] - mnew);
                s[i][0][r] = p0; s[i][1][r] = p1;
                float rs = p0 + p1;
                #pragma unroll
                for (int off = 1; off < 16; off <<= 1)
                    rs += __shfl_xor(rs, off, 32);
                lrow[i][r] = lrow[i][r] * corr + rs;
                #pragma unroll
                for (int jj = 0; jj < 4; ++jj) o[i][jj][r] *= corr;
            }

        // P (C-layout) -> LDS -> A-layout fragments (wave-private tile)
        unsigned short* lp = ldsP + wid * 1024;
        #pragma unroll
        for (int i = 0; i < 2; ++i)
            #pragma unroll
            for (int j = 0; j < 2; ++j)
                #pragma unroll
                for (int r = 0; r < 8; ++r)
                    lp[(i * 16 + r + half * 8) * 32 + j * 16 + l15] = f2bf(s[i][j][r]);

        ABFrag pf[2];
        #pragma unroll
        for (int i = 0; i < 2; ++i) {
            const uint4* p = (const uint4*)(lp + (i * 16 + l15) * 32 + koffA);
            pf[i].q[0] = p[0]; pf[i].q[1] = p[1];
        }

        // O += P @ V  (V^T rows are contiguous over keys)
        #pragma unroll
        for (int jj = 0; jj < 4; ++jj) {
            ABFrag vf;
            const uint4* p = (const uint4*)(vT + (size_t)(bh * HD + jj * 16 + l15) * LIN
                                               + l0 + koffB);
            vf.q[0] = p[0]; vf.q[1] = p[1];
            o[0][jj] = __builtin_amdgcn_wmma_f32_16x16x32_bf16(false, pf[0].v, false, vf.v, (short)0, o[0][jj], false, false);
            o[1][jj] = __builtin_amdgcn_wmma_f32_16x16x32_bf16(false, pf[1].v, false, vf.v, (short)0, o[1][jj], false, false);
        }
    }

    // normalize and store bf16 (B*NQ, D)
    #pragma unroll
    for (int i = 0; i < 2; ++i)
        #pragma unroll
        for (int r = 0; r < 8; ++r) {
            float inv = 1.0f / lrow[i][r];
            int row = b * NQ + q0 + i * 16 + r + half * 8;
            #pragma unroll
            for (int jj = 0; jj < 4; ++jj) {
                int col = hh * HD + jj * 16 + l15;
                oat[(size_t)row * D + col] = f2bf(o[i][jj][r] * inv);
            }
        }
}

// ---------------------------------------------------------------------------
// LayerNorm: one block (256 threads) per row, optional f32 / bf16 outputs.
// ---------------------------------------------------------------------------
__global__ __launch_bounds__(256)
void k_layernorm(const float* __restrict__ x,
                 const float* __restrict__ g, const float* __restrict__ bta,
                 float* __restrict__ outF, unsigned short* __restrict__ outB,
                 int ncols)
{
    __shared__ float red[20];
    const int row = blockIdx.x;
    const float* xr = x + (size_t)row * ncols;

    float s = 0.f, s2 = 0.f;
    for (int i = threadIdx.x; i < ncols; i += 256) {
        float v = xr[i]; s += v; s2 += v * v;
    }
    #pragma unroll
    for (int off = 16; off; off >>= 1) {
        s  += __shfl_xor(s,  off, 32);
        s2 += __shfl_xor(s2, off, 32);
    }
    int wid = threadIdx.x >> 5, lane = threadIdx.x & 31;
    if (lane == 0) { red[wid] = s; red[8 + wid] = s2; }
    __syncthreads();
    if (wid == 0) {
        s  = (lane < 8) ? red[lane]     : 0.f;
        s2 = (lane < 8) ? red[8 + lane] : 0.f;
        #pragma unroll
        for (int off = 4; off; off >>= 1) {
            s  += __shfl_xor(s,  off, 32);
            s2 += __shfl_xor(s2, off, 32);
        }
        if (lane == 0) {
            float m = s / ncols;
            red[16] = m;
            red[17] = rsqrtf(fmaxf(s2 / ncols - m * m, 0.f) + 1e-5f);
        }
    }
    __syncthreads();
    float mean = red[16], rstd = red[17];
    for (int i = threadIdx.x; i < ncols; i += 256) {
        float v = (xr[i] - mean) * rstd * g[i] + bta[i];
        size_t idx = (size_t)row * ncols + i;
        if (outF) outF[idx] = v;
        if (outB) outB[idx] = f2bf(v);
    }
}

// ---------------------------------------------------------------------------
// Small element-wise / data-movement kernels
// ---------------------------------------------------------------------------
__global__ void k_cvt_bf16(const float* __restrict__ s, unsigned short* __restrict__ d, int n) {
    int i = blockIdx.x * blockDim.x + threadIdx.x;
    if (i < n) d[i] = f2bf(s[i]);
}
__global__ void k_fill_f32(float* __restrict__ p, float v, int n) {
    int i = blockIdx.x * blockDim.x + threadIdx.x;
    if (i < n) p[i] = v;
}
// hq[b*NQ + i, :] = src[i, :]
__global__ void k_bcast_rows(const float* __restrict__ s, float* __restrict__ d, int n) {
    int i = blockIdx.x * blockDim.x + threadIdx.x;
    if (i < n) d[i] = s[i % (NQ * D)];
}
// vT[(b*H+h)*HD + d, l] = vh[(b*LIN+l)*D + h*HD + d]
__global__ void k_transpose_v(const unsigned short* __restrict__ vh, unsigned short* __restrict__ vT) {
    int i = blockIdx.x * blockDim.x + threadIdx.x;
    if (i >= Bz * H * HD * LIN) return;
    int l  = i % LIN;
    int t  = i / LIN;          // (b*H + h)*HD + d
    int d  = t % HD;
    int bh = t / HD;
    int hh = bh % H, b = bh / H;
    vT[i] = vh[(size_t)(b * LIN + l) * D + hh * HD + d];
}
__global__ void k_gelu_bf16(const float* __restrict__ s, unsigned short* __restrict__ d, int n) {
    int i = blockIdx.x * blockDim.x + threadIdx.x;
    if (i < n) {
        float x = s[i];
        d[i] = f2bf(0.5f * x * (1.f + erff(x * 0.70710678118654752f)));
    }
}
__global__ void k_softplus_inplace(float* __restrict__ p, int n) {
    int i = blockIdx.x * blockDim.x + threadIdx.x;
    if (i < n) {
        float v = p[i];
        p[i] = (v > 20.f) ? v : log1pf(__expf(v));
    }
}
// slice cvt: d[row*ncols + c] = bf16(s[row*lds + col0 + c])
__global__ void k_slice_bf16(const float* __restrict__ s, int lds_, int col0, int ncols,
                             unsigned short* __restrict__ d, int n) {
    int i = blockIdx.x * blockDim.x + threadIdx.x;
    if (i < n) {
        int c = i % ncols, row = i / ncols;
        d[i] = f2bf(s[(size_t)row * lds_ + col0 + c]);
    }
}
// causal depthwise conv (K=4) + bias + SiLU; xz rows are (b*NQ + t), width 2*DI
__global__ void k_conv_silu(const float* __restrict__ xz,
                            const float* __restrict__ cw, const float* __restrict__ cb,
                            float* __restrict__ xcF, unsigned short* __restrict__ xcB) {
    int i = blockIdx.x * blockDim.x + threadIdx.x;
    if (i >= Bz * NQ * DI) return;
    int d = i % DI, row = i / DI;
    int t = row % NQ;
    float acc = cb[d];
    #pragma unroll
    for (int k = 0; k < KCONV; ++k) {
        int tt = t - (KCONV - 1) + k;
        if (tt >= 0) acc += cw[d * KCONV + k] * xz[(size_t)(row - (KCONV - 1) + k) * (2 * DI) + d];
    }
    float s = acc / (1.f + __expf(-acc));
    xcF[i] = s; xcB[i] = f2bf(s);
}
// selective scan: one thread per (b,d); state DS=16 in registers
__global__ void k_mamba_scan(const float* __restrict__ delta, const float* __restrict__ dbl,
                             const float* __restrict__ xc, const float* __restrict__ Alog,
                             const float* __restrict__ Dp, float* __restrict__ y) {
    int i = blockIdx.x * blockDim.x + threadIdx.x;
    if (i >= Bz * DI) return;
    int d = i % DI, b = i / DI;
    float A[DS], h[DS];
    #pragma unroll
    for (int n = 0; n < DS; ++n) { A[n] = -__expf(Alog[d * DS + n]); h[n] = 0.f; }
    float Dd = Dp[d];
    for (int t = 0; t < NQ; ++t) {
        int row = b * NQ + t;
        float dt = delta[(size_t)row * DI + d];
        float xt = xc[(size_t)row * DI + d];
        float dx = dt * xt;
        const float* bm = dbl + (size_t)row * XPN + DTR;
        const float* cm = bm + DS;
        float acc = 0.f;
        #pragma unroll
        for (int n = 0; n < DS; ++n) {
            h[n] = __expf(dt * A[n]) * h[n] + dx * bm[n];
            acc += h[n] * cm[n];
        }
        y[(size_t)row * DI + d] = acc + Dd * xt;
    }
}
// yg = y * silu(z);  z = xz[:, DI + d]
__global__ void k_gate_bf16(const float* __restrict__ y, const float* __restrict__ xz,
                            unsigned short* __restrict__ yg, int n) {
    int i = blockIdx.x * blockDim.x + threadIdx.x;
    if (i < n) {
        int d = i % DI, row = i / DI;
        float z = xz[(size_t)row * (2 * DI) + DI + d];
        yg[i] = f2bf(y[i] * z / (1.f + __expf(-z)));
    }
}

// ---------------------------------------------------------------------------
// Host orchestration
// ---------------------------------------------------------------------------
static inline void* bump(char*& p, size_t bytes) {
    void* r = p; p += (bytes + 255) & ~(size_t)255; return r;
}
static inline void launch_gemm(hipStream_t st, const unsigned short* A, const unsigned short* W,
                               const float* bias, float* outF, unsigned short* outB,
                               int M, int N, int K, int acc) {
    dim3 g((N + 255) / 256, (M + 63) / 64);
    k_gemm_bf16<<<g, 256, 0, st>>>(A, W, bias, outF, outB, M, N, K, acc);
}
static inline void launch_cvt(hipStream_t st, const float* s, unsigned short* d, int n) {
    k_cvt_bf16<<<(n + 255) / 256, 256, 0, st>>>(s, d, n);
}

extern "C" void kernel_launch(void* const* d_in, const int* in_sizes, int n_in,
                              void* d_out, int out_size, void* d_ws, size_t ws_size,
                              hipStream_t stream) {
    (void)in_sizes; (void)n_in; (void)out_size; (void)ws_size;
    const float* x      = (const float*)d_in[0];
    const unsigned char* mask = (const unsigned char*)d_in[1];
    const float* W_in   = (const float*)d_in[2];
    const float* b_in   = (const float*)d_in[3];
    const float* latents= (const float*)d_in[4];
    const float* lnq_g  = (const float*)d_in[5];
    const float* lnq_b  = (const float*)d_in[6];
    const float* lnkv_g = (const float*)d_in[7];
    const float* lnkv_b = (const float*)d_in[8];
    const float* W_qkv  = (const float*)d_in[9];
    const float* b_qkv  = (const float*)d_in[10];
    const float* W_o    = (const float*)d_in[11];
    const float* b_o    = (const float*)d_in[12];
    const float* W1     = (const float*)d_in[13];
    const float* b1     = (const float*)d_in[14];
    const float* W2     = (const float*)d_in[15];
    const float* b2     = (const float*)d_in[16];
    const float* lnf_g  = (const float*)d_in[17];
    const float* lnf_b  = (const float*)d_in[18];
    const float* mb_ln_g= (const float*)d_in[19];
    const float* mb_ln_b= (const float*)d_in[20];
    const float* mb_Win = (const float*)d_in[21];
    const float* mb_cw  = (const float*)d_in[22];
    const float* mb_cb  = (const float*)d_in[23];
    const float* mb_xp  = (const float*)d_in[24];
    const float* mb_dtW = (const float*)d_in[25];
    const float* mb_dtb = (const float*)d_in[26];
    const float* mb_Alog= (const float*)d_in[27];
    const float* mb_D   = (const float*)d_in[28];
    const float* mb_Wout= (const float*)d_in[29];
    const float* enc_g  = (const float*)d_in[30];
    const float* enc_b  = (const float*)d_in[31];
    const float* out_g  = (const float*)d_in[32];
    const float* out_b  = (const float*)d_in[33];

    const int M1 = Bz * LIN;   // 8192
    const int MQ = Bz * NQ;    // 1024

    // ---- workspace bump allocation (~256 MB) ----
    char* p = (char*)d_ws;
    auto BF = [&](size_t n){ return (unsigned short*)bump(p, n * 2); };
    auto FP = [&](size_t n){ return (float*)bump(p, n * 4); };

    unsigned short* Win_b  = BF((size_t)D * DIN);
    unsigned short* Wqkv_b = BF((size_t)3 * D * D);
    unsigned short* Wo_b   = BF((size_t)D * D);
    unsigned short* W1_b   = BF((size_t)2 * D * D);
    unsigned short* W2_b   = BF((size_t)D * 2 * D);
    unsigned short* mWin_b = BF((size_t)NB * 2 * DI * D);
    unsigned short* mXp_b  = BF((size_t)NB * XPN * DI);
    unsigned short* mDtW_b = BF((size_t)NB * DI * DTR);
    unsigned short* mWo_b  = BF((size_t)NB * D * DI);

    unsigned short* x_b   = BF((size_t)M1 * DIN);
    float*          h_f   = FP((size_t)M1 * D);
    unsigned short* kv_b  = BF((size_t)M1 * D);
    float*          qln_f = FP((size_t)NQ * D);
    unsigned short* qln_b = BF((size_t)NQ * D);
    unsigned short* qh_b  = BF((size_t)NQ * D);
    unsigned short* kh_b  = BF((size_t)M1 * D);
    unsigned short* vh_b  = BF((size_t)M1 * D);
    unsigned short* vT_b  = BF((size_t)M1 * D);
    unsigned short* oat_b = BF((size_t)MQ * D);
    float*          hq_f  = FP((size_t)MQ * D);
    unsigned short* hq_b  = BF((size_t)MQ * D);
    float*          ff1_f = FP((size_t)MQ * 2 * D);
    unsigned short* g1_b  = BF((size_t)MQ * 2 * D);
    unsigned short* u_b   = BF((size_t)MQ * D);
    float*          xz_f  = FP((size_t)MQ * 2 * DI);
    float*          xc_f  = FP((size_t)MQ * DI);
    unsigned short* xc_b  = BF((size_t)MQ * DI);
    float*          dbl_f = FP((size_t)MQ * XPN);
    unsigned short* dtr_b = BF((size_t)MQ * DTR);
    float*          del_f = FP((size_t)MQ * DI);
    float*          y_f   = FP((size_t)MQ * DI);
    unsigned short* yg_b  = BF((size_t)MQ * DI);

    // ---- weight conversions (f32 -> bf16) ----
    launch_cvt(stream, W_in,   Win_b,  D * DIN);
    launch_cvt(stream, W_qkv,  Wqkv_b, 3 * D * D);
    launch_cvt(stream, W_o,    Wo_b,   D * D);
    launch_cvt(stream, W1,     W1_b,   2 * D * D);
    launch_cvt(stream, W2,     W2_b,   D * 2 * D);
    launch_cvt(stream, mb_Win, mWin_b, NB * 2 * DI * D);
    launch_cvt(stream, mb_xp,  mXp_b,  NB * XPN * DI);
    launch_cvt(stream, mb_dtW, mDtW_b, NB * DI * DTR);
    launch_cvt(stream, mb_Wout,mWo_b,  NB * D * DI);
    launch_cvt(stream, x,      x_b,    M1 * DIN);

    // ---- encoder front end ----
    launch_gemm(stream, x_b, Win_b, b_in, h_f, nullptr, M1, D, DIN, 0);          // h
    k_layernorm<<<NQ, 256, 0, stream>>>(latents, lnq_g, lnq_b, qln_f, qln_b, D); // q = LN(latents)
    k_layernorm<<<M1, 256, 0, stream>>>(h_f, lnkv_g, lnkv_b, nullptr, kv_b, D);  // kv = LN(h)

    launch_gemm(stream, qln_b, Wqkv_b,             b_qkv,         nullptr, qh_b, NQ, D, D, 0);
    launch_gemm(stream, kv_b,  Wqkv_b + (size_t)D * D,     b_qkv + D,     nullptr, kh_b, M1, D, D, 0);
    launch_gemm(stream, kv_b,  Wqkv_b + (size_t)2 * D * D, b_qkv + 2 * D, nullptr, vh_b, M1, D, D, 0);
    {
        int n = M1 * D;
        k_transpose_v<<<(n + 255) / 256, 256, 0, stream>>>(vh_b, vT_b);
    }
    k_flash_attn<<<Bz * H, 256, 0, stream>>>(qh_b, kh_b, vT_b, mask, oat_b);

    // hq = q (broadcast) + attn@W_o^T + b_o ; also emit bf16 copy for FFN1
    k_bcast_rows<<<(MQ * D + 255) / 256, 256, 0, stream>>>(qln_f, hq_f, MQ * D);
    launch_gemm(stream, oat_b, Wo_b, b_o, hq_f, hq_b, MQ, D, D, 1);

    // FFN: hq += GELU(hq@W1^T + b1)@W2^T + b2 ; then hq = LN(hq)
    launch_gemm(stream, hq_b, W1_b, b1, ff1_f, nullptr, MQ, 2 * D, D, 0);
    k_gelu_bf16<<<(MQ * 2 * D + 255) / 256, 256, 0, stream>>>(ff1_f, g1_b, MQ * 2 * D);
    launch_gemm(stream, g1_b, W2_b, b2, hq_f, nullptr, MQ, D, 2 * D, 1);
    k_layernorm<<<MQ, 256, 0, stream>>>(hq_f, lnf_g, lnf_b, hq_f, nullptr, D);

    // ---- 4 Mamba blocks ----
    for (int i = 0; i < NB; ++i) {
        k_layernorm<<<MQ, 256, 0, stream>>>(hq_f, mb_ln_g + i * D, mb_ln_b + i * D,
                                            nullptr, u_b, D);
        launch_gemm(stream, u_b, mWin_b + (size_t)i * 2 * DI * D, nullptr, xz_f, nullptr,
                    MQ, 2 * DI, D, 0);
        k_conv_silu<<<(MQ * DI + 255) / 256, 256, 0, stream>>>(
            xz_f, mb_cw + (size_t)i * DI * KCONV, mb_cb + (size_t)i * DI, xc_f, xc_b);
        launch_gemm(stream, xc_b, mXp_b + (size_t)i * XPN * DI, nullptr, dbl_f, nullptr,
                    MQ, XPN, DI, 0);
        k_slice_bf16<<<(MQ * DTR + 255) / 256, 256, 0, stream>>>(
            dbl_f, XPN, 0, DTR, dtr_b, MQ * DTR);
        launch_gemm(stream, dtr_b, mDtW_b + (size_t)i * DI * DTR, mb_dtb + (size_t)i * DI,
                    del_f, nullptr, MQ, DI, DTR, 0);
        k_softplus_inplace<<<(MQ * DI + 255) / 256, 256, 0, stream>>>(del_f, MQ * DI);
        k_mamba_scan<<<(Bz * DI + 255) / 256, 256, 0, stream>>>(
            del_f, dbl_f, xc_f, mb_Alog + (size_t)i * DI * DS, mb_D + (size_t)i * DI, y_f);
        k_gate_bf16<<<(MQ * DI + 255) / 256, 256, 0, stream>>>(y_f, xz_f, yg_b, MQ * DI);
        launch_gemm(stream, yg_b, mWo_b + (size_t)i * D * DI, nullptr, hq_f, nullptr,
                    MQ, D, DI, 1);  // hq += y@Wout^T
    }

    // ---- final LNs -> d_out; comp_mask (all true) appended as 1.0f ----
    float* out_f = (float*)d_out;
    k_layernorm<<<MQ, 256, 0, stream>>>(hq_f, enc_g, enc_b, hq_f, nullptr, D);
    k_layernorm<<<MQ, 256, 0, stream>>>(hq_f, out_g, out_b, out_f, nullptr, D);
    k_fill_f32<<<(MQ + 255) / 256, 256, 0, stream>>>(out_f + (size_t)MQ * D, 1.0f, MQ);
}
